// Encoder_61022895342133
// MI455X (gfx1250) — matrix operand order repaired and checked by reading, not compiled, not run on hardware
//
#include <hip/hip_runtime.h>
#include <hip/hip_bf16.h>

// ---------------------------------------------------------------------------
// Types for CDNA5 WMMA / TDM
// ---------------------------------------------------------------------------
typedef __attribute__((ext_vector_type(16))) __bf16 v16bf;
typedef __attribute__((ext_vector_type(8)))  float  v8f;
typedef __attribute__((ext_vector_type(4)))  unsigned int u32x4;
typedef __attribute__((ext_vector_type(4)))  int i32x4;
typedef __attribute__((ext_vector_type(8)))  int i32x8;

typedef unsigned short ushort_t;

// fp32 -> bf16 round-to-nearest-even (bit trick)
__device__ __forceinline__ ushort_t f2bf(float f) {
    unsigned int u = __float_as_uint(f);
    u += 0x7FFFu + ((u >> 16) & 1u);
    return (ushort_t)(u >> 16);
}
__device__ __forceinline__ float bf2f(ushort_t b) {
    return __uint_as_float(((unsigned int)b) << 16);
}
__device__ __forceinline__ float sigmoidf_(float x) {
    return 1.0f / (1.0f + __expf(-x));
}

// ---------------------------------------------------------------------------
// Problem constants (from reference setup_inputs)
// ---------------------------------------------------------------------------
#define T_STEPS 8192
#define F_DIM   256
#define HD      1024   // layer-1 hidden
#define E_DIM   512    // layer-2 hidden

// ---------------------------------------------------------------------------
// init: zero barrier + h double-buffers
// ---------------------------------------------------------------------------
__global__ __launch_bounds__(256) void init_kernel(unsigned int* bar,
                                                   float* hg1, float* hg2) {
    int tid = threadIdx.x;
    if (tid < 2) bar[tid] = 0u;
    for (int i = tid; i < 2 * HD; i += 256) hg1[i] = 0.0f;
    for (int i = tid; i < 2 * E_DIM; i += 256) hg2[i] = 0.0f;
}

// ---------------------------------------------------------------------------
// fp32 -> bf16 elementwise convert
// ---------------------------------------------------------------------------
__global__ __launch_bounds__(256) void convert_bf16_kernel(const float* __restrict__ in,
                                                           ushort_t* __restrict__ out,
                                                           long long n) {
    long long i = (long long)blockIdx.x * blockDim.x + threadIdx.x;
    long long stride = (long long)gridDim.x * blockDim.x;
    for (; i < n; i += stride) out[i] = f2bf(in[i]);
}

// ---------------------------------------------------------------------------
// WMMA bf16 GEMM with fused bias:
//   Y[M,N] (fp32) = Xbf[M,K] * Wbf[N,K]^T + b0[n] + b1[n]
// One 32x64 output strip per wave (2 row-tiles x 4 col-tiles, 8 v8f accs).
// K must be a multiple of 32; M multiple of 32; N multiple of 64.
// Fragment layouts per CDNA5 ISA 7.12.2:
//   A 16x32 bf16: lane m (0..15) row M=m; hi lanes K offset +8.
//     two contiguous 8-elem chunks at k0+hi*8 and k0+hi*8+16 (two 16B loads)
//   B 32x16 bf16: N=lane&15; same K order as A: chunks at k0+hi*8 and k0+hi*8+16 (two 16B loads)
//   C/D v8f: VGPR r -> M = r + 8*hi, N = lane&15
// ---------------------------------------------------------------------------
__global__ __launch_bounds__(128) void wmma_gemm_bias_kernel(
    const ushort_t* __restrict__ Xbf, const ushort_t* __restrict__ Wbf,
    const float* __restrict__ b0, const float* __restrict__ b1,
    float* __restrict__ Y, int M, int N, int K)
{
    const int wid  = (int)(blockIdx.x * (blockDim.x >> 5)) + ((int)threadIdx.x >> 5);
    const int lane = (int)threadIdx.x & 31;
    const int mt = M >> 5;                       // 32-row strips
    const int nt = N >> 6;                       // 64-col strips
    if (wid >= mt * nt) return;                  // wave-uniform exit
    const int tm = wid / nt, tn = wid % nt;
    const int row0 = tm << 5, col0 = tn << 6;

    const int r  = lane & 15;
    const int hi = lane >> 4;

    union Frag { v16bf v; u32x4 q[2]; };
    Frag A0, A1, B;
    v8f acc0[4] = {};   // rows row0..row0+15
    v8f acc1[4] = {};   // rows row0+16..row0+31

    const ushort_t* a0 = Xbf + (size_t)(row0 + r) * K + hi * 8;
    const ushort_t* a1 = a0 + (size_t)16 * K;
    const ushort_t* bb = Wbf + (size_t)(col0 + r) * K + hi * 8;

    for (int k0 = 0; k0 < K; k0 += 32) {
        A0.q[0] = *(const u32x4*)(a0 + k0);
        A0.q[1] = *(const u32x4*)(a0 + k0 + 16);
        A1.q[0] = *(const u32x4*)(a1 + k0);
        A1.q[1] = *(const u32x4*)(a1 + k0 + 16);
#pragma unroll
        for (int c = 0; c < 4; ++c) {
            const ushort_t* bc = bb + (size_t)(c << 4) * K;
            B.q[0] = *(const u32x4*)(bc + k0);
            B.q[1] = *(const u32x4*)(bc + k0 + 16);
            acc0[c] = __builtin_amdgcn_wmma_f32_16x16x32_bf16(
                false, A0.v, false, B.v, (short)0, acc0[c], false, false);
            acc1[c] = __builtin_amdgcn_wmma_f32_16x16x32_bf16(
                false, A1.v, false, B.v, (short)0, acc1[c], false, false);
        }
    }

#pragma unroll
    for (int c = 0; c < 4; ++c) {
        const int n = col0 + (c << 4) + r;
        const float bias = b0[n] + b1[n];
        float* y0 = Y + (size_t)(row0 + 8 * hi) * N + n;
        float* y1 = Y + (size_t)(row0 + 16 + 8 * hi) * N + n;
#pragma unroll
        for (int rr = 0; rr < 8; ++rr) {
            y0[(size_t)rr * N] = acc0[c][rr] + bias;
            y1[(size_t)rr * N] = acc1[c][rr] + bias;
        }
    }
}

// ---------------------------------------------------------------------------
// Global grid barrier (sense via generation counter). bar[0]=count, bar[1]=gen
// ---------------------------------------------------------------------------
__device__ __forceinline__ void grid_barrier(unsigned int* bar, int nwg) {
    __syncthreads();
    if (threadIdx.x == 0) {
        unsigned int g = __hip_atomic_load(bar + 1, __ATOMIC_RELAXED,
                                           __HIP_MEMORY_SCOPE_AGENT);
        unsigned int a = __hip_atomic_fetch_add(bar, 1u, __ATOMIC_ACQ_REL,
                                                __HIP_MEMORY_SCOPE_AGENT);
        if (a == (unsigned int)nwg - 1u) {
            __hip_atomic_store(bar, 0u, __ATOMIC_RELAXED, __HIP_MEMORY_SCOPE_AGENT);
            __hip_atomic_store(bar + 1, g + 1u, __ATOMIC_RELEASE,
                               __HIP_MEMORY_SCOPE_AGENT);
        } else {
            while (__hip_atomic_load(bar + 1, __ATOMIC_ACQUIRE,
                                     __HIP_MEMORY_SCOPE_AGENT) == g) {
                __builtin_amdgcn_s_sleep(2);
            }
        }
    }
    __syncthreads();
}

// ---------------------------------------------------------------------------
// TDM: stage the H-float hidden-state vector global -> LDS with one
// tensor_load_to_lds (1-D tile: tile_dim0 = H elements of 4 bytes).
// D# packing per CDNA5 ISA 8.3/8.4. Issued by wave 0 only (EXEC ignored by
// TDM); completion via s_wait_tensorcnt, then published by __syncthreads.
// This toolchain exposes the 6-arg builtin:
//   (u32x4 group0, i32x8 group1, i32x4 group2, i32x4 group3, i32x8, i32 cpol)
// ---------------------------------------------------------------------------
template <int H>
__device__ __forceinline__ void tdm_stage_h(const float* hin, float* hlds) {
    if (threadIdx.x < 32) {
        const unsigned int ldsAddr = (unsigned int)(unsigned long long)(void*)hlds;
        const unsigned long long ga = (unsigned long long)(const void*)hin;
        u32x4 g0;
        g0[0] = 1u;                                        // count=1, load, user D#
        g0[1] = ldsAddr;                                   // lds_addr
        g0[2] = (unsigned int)ga;                          // global_addr[31:0]
        g0[3] = (unsigned int)((ga >> 32) & 0x01FFFFFFu)   // global_addr[56:32]
                | (2u << 30);                              // type=2 (image)
        i32x8 g1;
        g1[0] = (int)(2u << 16);                           // data_size=4B, mask=0
        g1[1] = (int)((unsigned)H << 16);                  // tensor_dim0[15:0] @ bits63:48
        g1[2] = (int)(((unsigned)H >> 16) | (1u << 16));   // tensor_dim0 hi, tensor_dim1=1
        g1[3] = (int)((unsigned)H << 16);                  // tile_dim0 = H @ bits127:112
        g1[4] = 0;                                         // tile_dim1/2 unused
        g1[5] = (int)H;                                    // tensor_dim0_stride
        g1[6] = 0;
        g1[7] = 0;
        i32x4 z4 = {0, 0, 0, 0};
        i32x8 z8 = {0, 0, 0, 0, 0, 0, 0, 0};
        __builtin_amdgcn_tensor_load_to_lds(g0, g1, z4, z4, z8, 0);
        __builtin_amdgcn_s_wait_tensorcnt(0);
    }
    __syncthreads();
}

// ---------------------------------------------------------------------------
// Persistent sequential LSTM over T steps.
//   xg   : [T, 4H] precomputed input gates (fp32, i|f|g|o blocks of H)
//   whh  : [4H, H] recurrent weights (bf16, row-major, PyTorch layout)
//   hg   : [2, H]  double-buffered hidden state (fp32, buffer 0 zeroed)
//   hsBf : optional [T, H] bf16 hidden-state trace (feeds next layer's GEMM)
//   fin  : optional [H] fp32 final hidden output
// NWG workgroups x 256 threads, each WG owns H/NWG units; each unit's four
// gate dot-products are split over 256/(H/NWG) threads along K, reduced in
// LDS; the owner thread keeps (c,h) in registers for all T steps.
// One grid barrier per timestep; h staged into LDS by the TDM.
// ---------------------------------------------------------------------------
template <int H, int NWG>
__global__ __launch_bounds__(256) void lstm_seq_kernel(
    const float* __restrict__ xg, const ushort_t* __restrict__ whh,
    float* __restrict__ hg, ushort_t* __restrict__ hsBf,
    float* __restrict__ fin, int T, unsigned int* bar)
{
    constexpr int UPW = H / NWG;        // units per workgroup
    constexpr int TPU = 256 / UPW;      // threads per unit
    constexpr int KPT = H / TPU;        // K elements per thread

    const int tid    = (int)threadIdx.x;
    const int uLocal = tid / TPU;
    const int kpart  = tid % TPU;
    const int u      = (int)blockIdx.x * UPW + uLocal;
    const bool owner = (kpart == 0);
    const int kbase  = kpart * KPT;

    __shared__ float hlds[H];
    __shared__ float red[256 * 4];

    const ushort_t* w0 = whh + ((size_t)(0 * H + u)) * H + kbase;
    const ushort_t* w1 = whh + ((size_t)(1 * H + u)) * H + kbase;
    const ushort_t* w2 = whh + ((size_t)(2 * H + u)) * H + kbase;
    const ushort_t* w3 = whh + ((size_t)(3 * H + u)) * H + kbase;

    float c = 0.0f, h = 0.0f;

    for (int t = 0; t < T; ++t) {
        const float* hin  = hg + (size_t)(t & 1) * H;
        float*       hout = hg + (size_t)((t + 1) & 1) * H;

        tdm_stage_h<H>(hin, hlds);      // global h -> LDS via TDM + syncthreads

        float a0 = 0.f, a1 = 0.f, a2 = 0.f, a3 = 0.f;
#pragma unroll 8
        for (int k = 0; k < KPT; ++k) {
            const float hv = hlds[kbase + k];
            a0 += bf2f(w0[k]) * hv;
            a1 += bf2f(w1[k]) * hv;
            a2 += bf2f(w2[k]) * hv;
            a3 += bf2f(w3[k]) * hv;
        }
        red[tid * 4 + 0] = a0;
        red[tid * 4 + 1] = a1;
        red[tid * 4 + 2] = a2;
        red[tid * 4 + 3] = a3;
        __syncthreads();

        if (owner) {
            float g0 = 0.f, g1 = 0.f, g2 = 0.f, g3 = 0.f;
#pragma unroll
            for (int j = 0; j < TPU; ++j) {
                const int b = (tid + j) * 4;
                g0 += red[b + 0];
                g1 += red[b + 1];
                g2 += red[b + 2];
                g3 += red[b + 3];
            }
            const size_t xo = (size_t)t * (4 * H) + u;
            g0 += xg[xo + 0 * H];
            g1 += xg[xo + 1 * H];
            g2 += xg[xo + 2 * H];
            g3 += xg[xo + 3 * H];

            const float iG = sigmoidf_(g0);
            const float fG = sigmoidf_(g1);
            const float gG = tanhf(g2);
            const float oG = sigmoidf_(g3);
            c = fG * c + iG * gG;
            h = oG * tanhf(c);

            hout[u] = h;
            if (hsBf) hsBf[(size_t)t * H + u] = f2bf(h);
            __threadfence();   // make h (and trace) visible before the barrier
        }
        grid_barrier(bar, NWG);
    }

    if (owner && fin) fin[u] = h;
}

// ---------------------------------------------------------------------------
// Host side
// ---------------------------------------------------------------------------
extern "C" void kernel_launch(void* const* d_in, const int* in_sizes, int n_in,
                              void* d_out, int out_size, void* d_ws, size_t ws_size,
                              hipStream_t stream) {
    (void)in_sizes; (void)n_in; (void)out_size; (void)ws_size;

    const float* x     = (const float*)d_in[0];  // [8192, 256]
    const float* w_ih1 = (const float*)d_in[1];  // [4096, 256]
    const float* w_hh1 = (const float*)d_in[2];  // [4096, 1024]
    const float* b_ih1 = (const float*)d_in[3];  // [4096]
    const float* b_hh1 = (const float*)d_in[4];  // [4096]
    const float* w_ih2 = (const float*)d_in[5];  // [2048, 1024]
    const float* w_hh2 = (const float*)d_in[6];  // [2048, 512]
    const float* b_ih2 = (const float*)d_in[7];  // [2048]
    const float* b_hh2 = (const float*)d_in[8];  // [2048]
    float* out = (float*)d_out;                  // [512]

    char* ws = (char*)d_ws;
    size_t off = 0;
    auto alloc = [&](size_t bytes) -> char* {
        char* p = ws + off;
        off = (off + bytes + 255) & ~(size_t)255;
        return p;
    };

    unsigned int* bar   = (unsigned int*)alloc(256);
    ushort_t* xbf       = (ushort_t*)alloc((size_t)T_STEPS * F_DIM * 2);
    ushort_t* wih1bf    = (ushort_t*)alloc((size_t)4 * HD * F_DIM * 2);
    ushort_t* whh1bf    = (ushort_t*)alloc((size_t)4 * HD * HD * 2);
    ushort_t* wih2bf    = (ushort_t*)alloc((size_t)4 * E_DIM * HD * 2);
    ushort_t* whh2bf    = (ushort_t*)alloc((size_t)4 * E_DIM * E_DIM * 2);
    float*    xg1       = (float*)alloc((size_t)T_STEPS * 4 * HD * 4);
    ushort_t* hs1bf     = (ushort_t*)alloc((size_t)T_STEPS * HD * 2);
    float*    xg2       = (float*)alloc((size_t)T_STEPS * 4 * E_DIM * 4);
    float*    hg1       = (float*)alloc((size_t)2 * HD * 4);
    float*    hg2       = (float*)alloc((size_t)2 * E_DIM * 4);

    // 0) zero barrier + hidden-state double buffers
    init_kernel<<<1, 256, 0, stream>>>(bar, hg1, hg2);

    // 1) fp32 -> bf16 conversions
    convert_bf16_kernel<<<4096, 256, 0, stream>>>(x,     xbf,    (long long)T_STEPS * F_DIM);
    convert_bf16_kernel<<<4096, 256, 0, stream>>>(w_ih1, wih1bf, (long long)4 * HD * F_DIM);
    convert_bf16_kernel<<<4096, 256, 0, stream>>>(w_hh1, whh1bf, (long long)4 * HD * HD);
    convert_bf16_kernel<<<4096, 256, 0, stream>>>(w_ih2, wih2bf, (long long)4 * E_DIM * HD);
    convert_bf16_kernel<<<4096, 256, 0, stream>>>(w_hh2, whh2bf, (long long)4 * E_DIM * E_DIM);

    // 2) xg1 = x @ w_ih1^T + b_ih1 + b_hh1   [8192 x 4096], K=256
    {
        const int M = T_STEPS, N = 4 * HD, K = F_DIM;
        const int waves = (M / 32) * (N / 64);
        wmma_gemm_bias_kernel<<<(waves + 3) / 4, 128, 0, stream>>>(
            xbf, wih1bf, b_ih1, b_hh1, xg1, M, N, K);
    }

    // 3) layer-1 recurrence -> hs1 (bf16 trace)
    lstm_seq_kernel<HD, 32><<<32, 256, 0, stream>>>(
        xg1, whh1bf, hg1, hs1bf, nullptr, T_STEPS, bar);

    // 4) xg2 = hs1 @ w_ih2^T + b_ih2 + b_hh2   [8192 x 2048], K=1024
    {
        const int M = T_STEPS, N = 4 * E_DIM, K = HD;
        const int waves = (M / 32) * (N / 64);
        wmma_gemm_bias_kernel<<<(waves + 3) / 4, 128, 0, stream>>>(
            hs1bf, wih2bf, b_ih2, b_hh2, xg2, M, N, K);
    }

    // 5) layer-2 recurrence -> final hidden straight into d_out
    lstm_seq_kernel<E_DIM, 32><<<32, 256, 0, stream>>>(
        xg2, whh2bf, hg2, nullptr, out, T_STEPS, bar);
}
